// MultiheadAttentionNoFlash_85925115723827
// MI455X (gfx1250) — compile-verified, matrix operand order checked by reading, not run on hardware
//
#include <hip/hip_runtime.h>

typedef __bf16 bf16;
typedef __attribute__((ext_vector_type(16))) __bf16 v16bf;
typedef __attribute__((ext_vector_type(8)))  __bf16 v8bf;
typedef __attribute__((ext_vector_type(8)))  float  v8f;

#define DIMW   1024
#define NHEAD  16
#define HD     64
#define BSZ    2
#define SLEN   2048
#define MROWS  (BSZ * SLEN)   // 4096
#define KD     1024           // contraction dim of every linear layer

// ---------- helpers ----------

// Native f32 -> bf16 (RNE); backend picks the hardware convert op.
__device__ __forceinline__ bf16 f2bf(float f) { return (bf16)f; }

// Fixed SWAPX16 butterfly across lane halves: ds_swizzle group-of-32 encoding,
// xor_mask=0x10, or_mask=0, and_mask=0x1f  ->  offset = 0x401F. No LDS memory
// traffic, no generic-shuffle index math.
__device__ __forceinline__ float swap16(float x) {
    int i = __builtin_amdgcn_ds_swizzle(__builtin_bit_cast(int, x), 0x401F);
    return __builtin_bit_cast(float, i);
}

// Load a 16x32 (MxK) bf16 A-fragment (also used for B-frags: B^T rows) from
// row-major memory. Per ISA: lanes 0-15 hold row (lane), K = [0..7,16..23];
// lanes 16-31 hold row (lane-16), K = [8..15,24..31]. Two 16B chunks per lane.
__device__ __forceinline__ v16bf load_frag(const bf16* base, int ld, int lane) {
    int r  = lane & 15;
    int kb = (lane >> 4) << 3;          // 0 or 8
    const bf16* p = base + (size_t)r * ld + kb;
    union { v16bf v; v8bf h[2]; } u;
    u.h[0] = *(const v8bf*)(p);
    u.h[1] = *(const v8bf*)(p + 16);
    return u.v;
}

__device__ __forceinline__ v8f wmma_bf16(v16bf a, v16bf b, v8f c) {
    return __builtin_amdgcn_wmma_f32_16x16x32_bf16(
        /*neg_a=*/false, a, /*neg_b=*/false, b,
        /*c_mod=*/(short)0, c, /*reuse_a=*/false, /*reuse_b=*/false);
}

// Async global -> LDS copy of 16 bytes per lane (CDNA5 ASYNCcnt path; no VGPR
// data movement). ldsoff = byte offset inside the workgroup LDS allocation
// (low 32 bits of the flat shared pointer, per the LDS aperture rules).
__device__ __forceinline__ void async_ld_lds_b128(unsigned ldsoff, const void* gaddr) {
    asm volatile("global_load_async_to_lds_b128 %0, %1, off"
                 :: "v"(ldsoff), "v"((unsigned long long)(size_t)gaddr)
                 : "memory");
}

__device__ __forceinline__ void wait_asynccnt0() {
    asm volatile("s_wait_asynccnt 0" ::: "memory");
}

// ---------- fp32 -> bf16 convert ----------

__global__ void cvt_f32_bf16(const float* __restrict__ in, bf16* __restrict__ out, int n) {
    int i = blockIdx.x * blockDim.x + threadIdx.x;
    int stride = gridDim.x * blockDim.x;
    for (; i < n; i += stride) out[i] = f2bf(in[i]);
}

// ---------- NT GEMM: C[M,N] = alpha * A[M,KD] @ B[N,KD]^T ----------
// 256 threads = 8 waves; block tile 128x128; each wave: 32x64 via 2x4 WMMA tiles.

template <bool OUT_F32>
__global__ __launch_bounds__(256) void gemm_nt(
    const bf16* __restrict__ A, const bf16* __restrict__ Bm,
    void* __restrict__ Cout, int N, float alpha)
{
    int lane = threadIdx.x & 31;
    int w    = threadIdx.x >> 5;
    int m0 = blockIdx.x * 128 + (w & 3) * 32;
    int n0 = blockIdx.y * 128 + (w >> 2) * 64;

    const bf16* Abase = A  + (size_t)m0 * KD;
    const bf16* Bbase = Bm + (size_t)n0 * KD;

    v8f acc[2][4] = {};

#pragma unroll 2
    for (int k0 = 0; k0 < KD; k0 += 32) {
        v16bf af[2], bfr[4];
#pragma unroll
        for (int i = 0; i < 2; ++i)
            af[i] = load_frag(Abase + (size_t)(16 * i) * KD + k0, KD, lane);
#pragma unroll
        for (int j = 0; j < 4; ++j)
            bfr[j] = load_frag(Bbase + (size_t)(16 * j) * KD + k0, KD, lane);

        if (k0 + 32 < KD) {
            __builtin_prefetch(Abase + (size_t)(lane & 15) * KD + k0 + 32, 0, 3);
            __builtin_prefetch(Bbase + (size_t)(lane & 15) * KD + k0 + 32, 0, 3);
        }

#pragma unroll
        for (int i = 0; i < 2; ++i)
#pragma unroll
            for (int j = 0; j < 4; ++j)
                acc[i][j] = wmma_bf16(af[i], bfr[j], acc[i][j]);
    }

    int half = lane >> 4, col = lane & 15;
#pragma unroll
    for (int i = 0; i < 2; ++i) {
#pragma unroll
        for (int r = 0; r < 8; ++r) {
            size_t row = (size_t)(m0 + 16 * i + r + 8 * half) * N + n0 + col;
#pragma unroll
            for (int j = 0; j < 4; ++j) {
                float v = acc[i][j][r] * alpha;
                if (OUT_F32) ((float*)Cout)[row + 16 * j] = v;
                else         ((bf16*) Cout)[row + 16 * j] = f2bf(v);
            }
        }
    }
}

// ---------- fused attention, transposed-scores flash softmax ----------
// grid = B*H*(S/64) blocks, 128 threads = 4 waves; each wave owns 16 q-rows.
// Scores are computed transposed: S^T(keys x q) = K_tile x Q^T, so each lane
// owns one q column -> softmax stats are per-lane scalars (1 swizzle/block),
// and exp'd probs pack directly into the P^T B-fragment (no LDS round-trip).
// K tile is staged with GLOBAL_LOAD_ASYNC_TO_LDS (ASYNCcnt); V goes through
// VGPRs because it must be transposed on the way into LDS.
// Q was pre-scaled by SCALE=0.125 in the projection GEMM.

__global__ __launch_bounds__(128) void attn_fwd(
    const bf16* __restrict__ Q, const bf16* __restrict__ Kt,
    const bf16* __restrict__ V, bf16* __restrict__ Ctx)
{
    __shared__ __align__(16) bf16 lK[32 * 64];      // K tile, row-major (key, hd)
    __shared__ __align__(16) bf16 lVT[64 * 32];     // V tile transposed (hd, key)

    int tid  = threadIdx.x;
    int lane = tid & 31;
    int w    = tid >> 5;

    const int nqb = SLEN / 64;                       // 32 q-blocks per (b,h)
    int qb = blockIdx.x % nqb;
    int h  = (blockIdx.x / nqb) % NHEAD;
    int b  = blockIdx.x / (nqb * NHEAD);
    int q0 = qb * 64 + w * 16;

    // Q rows in A-layout; identical register image serves as the B-fragment
    // of Q^T in the transposed score WMMA.
    const bf16* Qp = Q + (size_t)(b * SLEN + q0) * DIMW + h * HD;
    v16bf aq0 = load_frag(Qp,      DIMW, lane);      // hd 0..31
    v16bf aq1 = load_frag(Qp + 32, DIMW, lane);      // hd 32..63

    v8f o[4] = {};                 // O^T tiles: (hd 16-chunk) x (q)
    float mrun = -3.0e38f;         // per-lane: q = lane&15 (replicated in halves)
    float lrun = 0.0f;

    int lrow = tid >> 2;          // 0..31 : local key row
    int lcol = (tid & 3) * 16;    // 0,16,32,48 : hd column chunk

    // LDS byte offsets for this thread's two K-tile chunks
    unsigned lkoff0 = (unsigned)(size_t)(lK + lrow * 64 + lcol);
    unsigned lkoff1 = lkoff0 + 16;

    for (int kb = 0; kb < SLEN; kb += 32) {
        // --- K tile: async global->LDS, no VGPR staging ---
        const bf16* kp = Kt + (size_t)(b * SLEN + kb + lrow) * DIMW + h * HD + lcol;
        async_ld_lds_b128(lkoff0, kp);
        async_ld_lds_b128(lkoff1, kp + 8);

        // --- V tile: through VGPRs, transposed into LDS ---
        const bf16* vp = V + (size_t)(b * SLEN + kb + lrow) * DIMW + h * HD + lcol;
        union { v8bf v; bf16 e[8]; } v0, v1;
        v0.v = *(const v8bf*)(vp);
        v1.v = *(const v8bf*)(vp + 8);
#pragma unroll
        for (int j = 0; j < 8; ++j) {
            lVT[(lcol + j)     * 32 + lrow] = v0.e[j];
            lVT[(lcol + 8 + j) * 32 + lrow] = v1.e[j];
        }

        if (kb + 32 < SLEN) {
            __builtin_prefetch(vp + 32 * DIMW, 0, 3);
        }
        wait_asynccnt0();       // our async K chunks landed in LDS
        __syncthreads();        // + everyone's ds stores / asyncs visible

        // --- S^T(32 keys x 16 q): two 16-key M-tiles, hd split 32+32 ---
        v8f st[2];
#pragma unroll
        for (int mt = 0; mt < 2; ++mt) {
            v16bf ak0 = load_frag(lK + mt * 16 * 64,      64, lane);
            v16bf ak1 = load_frag(lK + mt * 16 * 64 + 32, 64, lane);
            v8f z = {};
            st[mt] = wmma_bf16(ak1, aq1, wmma_bf16(ak0, aq0, z));
        }
        // C-layout of S^T: lane n (0-15) holds keys mt*16+r   for q=n,
        //                  lane n+16    holds keys mt*16+r+8  for q=n.

        // --- online softmax: per-lane scalar stats, one cross-half swizzle ---
        float mx = st[0][0];
#pragma unroll
        for (int r = 1; r < 8; ++r) mx = fmaxf(mx, st[0][r]);
#pragma unroll
        for (int r = 0; r < 8; ++r) mx = fmaxf(mx, st[1][r]);
        mx = fmaxf(mx, swap16(mx));
        float mnew = fmaxf(mrun, mx);

        float p[16];
        float sum = 0.f;
#pragma unroll
        for (int r = 0; r < 8; ++r) { p[r]     = __expf(st[0][r] - mnew); sum += p[r]; }
#pragma unroll
        for (int r = 0; r < 8; ++r) { p[8 + r] = __expf(st[1][r] - mnew); sum += p[8 + r]; }
        sum += swap16(sum);

        float corr = __expf(mrun - mnew);
        lrun = lrun * corr + sum;
        mrun = mnew;
#pragma unroll
        for (int t = 0; t < 4; ++t) o[t] *= corr;

        // --- pack P^T directly into its B-fragment (keys x q, K-split by half) ---
        v16bf pf;
#pragma unroll
        for (int i = 0; i < 16; ++i) pf[i] = f2bf(p[i]);

        // --- O^T += V^T(hd x keys) @ P^T(keys x q) ---
#pragma unroll
        for (int t = 0; t < 4; ++t) {
            v16bf av = load_frag(lVT + t * 16 * 32, 32, lane);
            o[t] = wmma_bf16(av, pf, o[t]);
        }
        __syncthreads();   // protect lK/lVT before next iteration overwrites
    }

    // --- epilogue: normalize; per lane one q-row, contiguous hd -> b128 stores ---
    float inv = 1.0f / lrun;
    int qrow = lane & 15, half = lane >> 4;
    bf16* dst = Ctx + (size_t)(b * SLEN + q0 + qrow) * DIMW + h * HD + 8 * half;
#pragma unroll
    for (int t = 0; t < 4; ++t) {
        union { v8bf v; bf16 e[8]; } pk;
#pragma unroll
        for (int r = 0; r < 8; ++r) pk.e[r] = f2bf(o[t][r] * inv);
        *(v8bf*)(dst + 16 * t) = pk.v;
    }
}

// ---------- host-side orchestration ----------

extern "C" void kernel_launch(void* const* d_in, const int* in_sizes, int n_in,
                              void* d_out, int out_size, void* d_ws, size_t ws_size,
                              hipStream_t stream) {
    (void)in_sizes; (void)n_in; (void)out_size; (void)ws_size;
    const float* q  = (const float*)d_in[0];
    const float* Wq = (const float*)d_in[1];
    const float* Wk = (const float*)d_in[2];
    const float* Wv = (const float*)d_in[3];
    const float* Wo = (const float*)d_in[4];
    float* out = (float*)d_out;

    char* wsb = (char*)d_ws;
    size_t off = 0;
    auto take = [&](size_t elems) -> bf16* {
        bf16* p = (bf16*)(wsb + off);
        off += elems * sizeof(bf16);
        return p;
    };
    const size_t XE = (size_t)MROWS * DIMW;   // 4096*1024
    const size_t WE = (size_t)DIMW * DIMW;    // 1024*1024
    bf16* Xb  = take(XE);
    bf16* Wqb = take(WE);
    bf16* Wkb = take(WE);
    bf16* Wvb = take(WE);
    bf16* Wob = take(WE);
    bf16* Qb  = take(XE);
    bf16* Kb  = take(XE);
    bf16* Vb  = take(XE);
    bf16* Cxb = take(XE);

    // 1) convert inputs to bf16
    cvt_f32_bf16<<<1024, 256, 0, stream>>>(q,  Xb,  (int)XE);
    cvt_f32_bf16<<<512,  256, 0, stream>>>(Wq, Wqb, (int)WE);
    cvt_f32_bf16<<<512,  256, 0, stream>>>(Wk, Wkb, (int)WE);
    cvt_f32_bf16<<<512,  256, 0, stream>>>(Wv, Wvb, (int)WE);
    cvt_f32_bf16<<<512,  256, 0, stream>>>(Wo, Wob, (int)WE);

    // 2) projections (Q pre-scaled by softmax scale 1/sqrt(64))
    dim3 gg(MROWS / 128, DIMW / 128), gb(256);
    gemm_nt<false><<<gg, gb, 0, stream>>>(Xb, Wqb, Qb, DIMW, 0.125f);
    gemm_nt<false><<<gg, gb, 0, stream>>>(Xb, Wkb, Kb, DIMW, 1.0f);
    gemm_nt<false><<<gg, gb, 0, stream>>>(Xb, Wvb, Vb, DIMW, 1.0f);

    // 3) fused attention -> head-merged context
    attn_fwd<<<BSZ * NHEAD * (SLEN / 64), 128, 0, stream>>>(Qb, Kb, Vb, Cxb);

    // 4) output projection, f32 result
    gemm_nt<true><<<gg, gb, 0, stream>>>(Cxb, Wob, out, DIMW, 1.0f);
}
